// NeuralODE_30605936951975
// MI455X (gfx1250) — matrix operand order repaired and checked by reading, not compiled, not run on hardware
//
#include <hip/hip_runtime.h>
#include <hip/hip_bf16.h>

typedef __attribute__((ext_vector_type(16))) _Float16 v16h;
typedef __attribute__((ext_vector_type(8)))  _Float16 v8h;
typedef __attribute__((ext_vector_type(8)))  float    v8f;

#define DIN    64
#define HDIM   128
#define H2DIM  256
#define DOUT   64
#define NSTEPS 20
#define DT     (1.0f / 20.0f)

// f16 workspace layout (element offsets): weights transposed to [N][K] (K contiguous)
#define OFF_ENC 0        // 128 x 64
#define OFF_W1  8192     // 256 x 128
#define OFF_W2  40960    // 256 x 256
#define OFF_W3  106496   // 128 x 256
#define OFF_DEC 139264   // 64 x 128
#define WS_HALVES 147456

// LDS layout (halves): [ W2: 65536 ][ per wave: buf0 4096 | buf1 4096 | sum 4096 ] x 8
#define W2_LDS_H   65536
#define WAVE_LDS_H 12288
#define WAVES_PER_BLOCK 8
#define TOTAL_LDS_H (W2_LDS_H + WAVES_PER_BLOCK * WAVE_LDS_H)   // 163840 h = 320 KB

// ---------- native TRANS-pipe tanh (v_tanh_f32 confirmed in asm) ----------
__device__ __forceinline__ float fast_tanh(float x) {
#if __has_builtin(__builtin_amdgcn_tanhf)
    return __builtin_amdgcn_tanhf(x);
#else
    float e = __builtin_amdgcn_exp2f(x * 2.8853900817779268f);  // 2/ln(2)
    return 1.0f - 2.0f * __builtin_amdgcn_rcpf(e + 1.0f);
#endif
}

// ---------- fragment loads (wave32 WMMA layouts, CDNA5 ISA 7.12.2) ----------

// A fragment: 16x32 f16 tile from LDS activation buffer (row-major, row stride K).
__device__ __forceinline__ v16h load_a(const _Float16* p, int K, int kb, int lane) {
    int m = lane & 15;
    int g = lane >> 4;
    const _Float16* q = p + m * K + kb * 32 + g * 16;
    v8h lo = *(const v8h*)q;          // ds_load_b128
    v8h hi = *(const v8h*)(q + 8);    // ds_load_b128
    return __builtin_shufflevector(lo, hi, 0,1,2,3,4,5,6,7,8,9,10,11,12,13,14,15);
}

// B fragment: 32x16 f16 tile from pre-transposed weight Wt[N][K] (global L2/WGP$ or LDS).
__device__ __forceinline__ v16h load_b(const _Float16* Wt, int K, int nt, int kb, int lane) {
    int n  = nt * 16 + (lane & 15);
    int k0 = kb * 32 + (lane >> 4) * 16;
    return *(const v16h*)(Wt + (size_t)n * K + k0);  // 32B contiguous
}

// One 16x16 output tile. C starts at inline 0 (backend emits WMMA with literal-0 SRC2);
// bias added in epilogue. Unroll capped at 4 to bound in-flight fragment registers.
template<int K>
__device__ __forceinline__ v8f gemm_tile(const _Float16* aLds, const _Float16* Wt,
                                         const float* bias, int nt, int lane) {
    float bv = bias[nt * 16 + (lane & 15)];
    v8f acc = {0.f, 0.f, 0.f, 0.f, 0.f, 0.f, 0.f, 0.f};
    #pragma unroll 4
    for (int kb = 0; kb < K / 32; ++kb) {
        v16h a = load_a(aLds, K, kb, lane);
        v16h b = load_b(Wt, K, nt, kb, lane);
        acc = __builtin_amdgcn_wmma_f32_16x16x32_f16(false, a, false, b,
                                                     (short)0, acc, false, false);
    }
    #pragma unroll
    for (int i = 0; i < 8; ++i) acc[i] += bv;
    return acc;
}

// Store C fragment (optionally tanh'd) as f16 into LDS row-major [16 x N].
__device__ __forceinline__ void store_tile(_Float16* out, int N, int nt, v8f v,
                                           int lane, bool dotanh) {
    int col = nt * 16 + (lane & 15);
    int r0  = (lane >> 4) * 8;
    #pragma unroll
    for (int i = 0; i < 8; ++i) {
        float x = v[i];
        if (dotanh) x = fast_tanh(x);
        out[(r0 + i) * N + col] = (_Float16)x;
    }
}

// ---------- ODE dynamics f(h), fused RK4 epilogue ----------
// MODE 0 (k1):     sumb  = k (plain ds_store_b32, no pre-zeroing); write h + cdt*k to bufT.
// MODE 1 (k2,k3):  sumb += wk*k via ds_add_f32;                    write h + cdt*k to bufT.
// MODE 2 (k4):     h += (DT/6)*(sumb + k); write new f16 state to bufT.
// Per-wave LDS regions -> no barriers (DS ops are in-order within a wave).
template<int MODE>
__device__ __forceinline__ void ode_f(_Float16* bufS, _Float16* bufT, float* sumb,
                                      const _Float16* Wt1, const float* b1,
                                      const _Float16* W2L, const float* b2,
                                      const _Float16* Wt3, const float* b3,
                                      v8f* h, float wk, float cdt, int lane) {
    #pragma unroll 1
    for (int nt = 0; nt < 16; ++nt) {           // 128 -> 256, tanh (W1 streamed, fits WGP$)
        v8f t = gemm_tile<HDIM>(bufS, Wt1, b1, nt, lane);
        store_tile(bufT, H2DIM, nt, t, lane, true);
    }
    #pragma unroll 1
    for (int nt = 0; nt < 16; ++nt) {           // 256 -> 256, tanh (W2 LDS-resident)
        v8f t = gemm_tile<H2DIM>(bufT, W2L, b2, nt, lane);
        store_tile(bufS, H2DIM, nt, t, lane, true);
    }
    const int col0 = lane & 15;
    const int r0   = (lane >> 4) * 8;
    #pragma unroll
    for (int nt = 0; nt < 8; ++nt) {            // 256 -> 128, linear (W3 streamed, fits WGP$)
        v8f k = gemm_tile<H2DIM>(bufS, Wt3, b3, nt, lane);
        int col = nt * 16 + col0;
        if (MODE == 0) {
            #pragma unroll
            for (int i = 0; i < 8; ++i)
                sumb[(r0 + i) * HDIM + col] = k[i];                  // init sum (k1, wk=1)
            v8f nx = h[nt] + k * cdt;
            store_tile(bufT, HDIM, nt, nx, lane, false);
        } else if (MODE == 1) {
            #pragma unroll
            for (int i = 0; i < 8; ++i)
                atomicAdd(&sumb[(r0 + i) * HDIM + col], k[i] * wk);  // ds_add_f32 no-rtn
            v8f nx = h[nt] + k * cdt;
            store_tile(bufT, HDIM, nt, nx, lane, false);
        } else {
            #pragma unroll
            for (int i = 0; i < 8; ++i) {
                float s  = sumb[(r0 + i) * HDIM + col];
                float hn = h[nt][i] + (DT / 6.0f) * (s + k[i]);
                h[nt][i] = hn;
                bufT[(r0 + i) * HDIM + col] = (_Float16)hn;          // state for next step
            }
        }
    }
}

// ---------- main persistent kernel: one wave owns 16 batch rows end-to-end ----------
extern "C" __global__ void __launch_bounds__(256)
node_kernel(const float* __restrict__ x, const _Float16* __restrict__ ws,
            const float* __restrict__ b_enc, const float* __restrict__ b1,
            const float* __restrict__ b2, const float* __restrict__ b3,
            const float* __restrict__ b_dec, float* __restrict__ out) {
    extern __shared__ _Float16 smem[];
    const int lane = threadIdx.x & 31;
    const int wib  = threadIdx.x >> 5;

    _Float16* W2L  = smem;                              // 256x256 f16, block-shared
    _Float16* base = smem + W2_LDS_H + wib * WAVE_LDS_H;
    _Float16* buf0 = base;
    _Float16* buf1 = base + 4096;
    float*    sumb = (float*)(base + 8192);             // 16 x 128 fp32 RK4 accumulator

    const int wave    = blockIdx.x * WAVES_PER_BLOCK + wib;
    const int rowbase = wave * 16;

    const _Float16* Wt_enc = ws + OFF_ENC;
    const _Float16* Wt1    = ws + OFF_W1;
    const _Float16* Wt3    = ws + OFF_W3;
    const _Float16* Wt_dec = ws + OFF_DEC;

    // stage W2 (128 KB) into LDS once per block: all 8 waves share it afterwards
    {
        const v8h* src = (const v8h*)(ws + OFF_W2);
        v8h*       dst = (v8h*)W2L;
        for (int i = threadIdx.x; i < W2_LDS_H / 8; i += 256) dst[i] = src[i];
    }
    __syncthreads();

    // encoder: load x tile (16 x 64) to LDS as f16, GEMM + tanh -> h registers
    const float* xg = x + (size_t)rowbase * DIN;
    for (int i = lane; i < 16 * DIN; i += 32) buf0[i] = (_Float16)xg[i];

    v8f h[8];
    #pragma unroll
    for (int nt = 0; nt < 8; ++nt) {
        v8f t = gemm_tile<DIN>(buf0, Wt_enc, b_enc, nt, lane);
        #pragma unroll
        for (int i = 0; i < 8; ++i) t[i] = fast_tanh(t[i]);
        h[nt] = t;
        store_tile(buf1, HDIM, nt, t, lane, false);   // state lives in buf1 first
    }

    #pragma unroll 1
    for (int s = 0; s < NSTEPS; ++s) {
        ode_f<0>(buf1, buf0, sumb, Wt1, b1, W2L, b2, Wt3, b3, h, 1.0f, 0.5f * DT, lane);
        ode_f<1>(buf0, buf1, sumb, Wt1, b1, W2L, b2, Wt3, b3, h, 2.0f, 0.5f * DT, lane);
        ode_f<1>(buf1, buf0, sumb, Wt1, b1, W2L, b2, Wt3, b3, h, 2.0f, DT,        lane);
        ode_f<2>(buf0, buf1, sumb, Wt1, b1, W2L, b2, Wt3, b3, h, 1.0f, 0.0f,      lane);
        // h updated in-place; new f16 state already in buf1 for next step / decoder
    }

    // decoder: hT (buf1) x W_dec + b_dec -> fp32 out
    #pragma unroll
    for (int nt = 0; nt < 4; ++nt) {
        v8f o = gemm_tile<HDIM>(buf1, Wt_dec, b_dec, nt, lane);
        int col = nt * 16 + (lane & 15);
        int r0  = (lane >> 4) * 8;
        #pragma unroll
        for (int i = 0; i < 8; ++i)
            out[(size_t)(rowbase + r0 + i) * DOUT + col] = o[i];
    }
}

// ---------- weight prep: transpose fp32 [K][N] -> f16 [N][K] (K contiguous) ----------
__device__ __forceinline__ void tc(const float* src, int K, int N, _Float16* dst, int idx) {
    int n = idx / K, k = idx - n * K;
    dst[(size_t)n * K + k] = (_Float16)src[(size_t)k * N + n];
}

extern "C" __global__ void __launch_bounds__(256)
prep_kernel(const float* We, const float* W1, const float* W2, const float* W3,
            const float* Wd, _Float16* ws) {
    int t = blockIdx.x * 256 + threadIdx.x;
    if      (t < 8192)   tc(We, 64,  128, ws + OFF_ENC, t);
    else if (t < 40960)  tc(W1, 128, 256, ws + OFF_W1,  t - 8192);
    else if (t < 106496) tc(W2, 256, 256, ws + OFF_W2,  t - 40960);
    else if (t < 139264) tc(W3, 256, 128, ws + OFF_W3,  t - 106496);
    else if (t < 147456) tc(Wd, 128, 64,  ws + OFF_DEC, t - 139264);
}

extern "C" void kernel_launch(void* const* d_in, const int* in_sizes, int n_in,
                              void* d_out, int out_size, void* d_ws, size_t ws_size,
                              hipStream_t stream) {
    (void)in_sizes; (void)n_in; (void)out_size; (void)ws_size;
    const float* x     = (const float*)d_in[0];
    const float* W_enc = (const float*)d_in[1];
    const float* b_enc = (const float*)d_in[2];
    const float* W1    = (const float*)d_in[3];
    const float* b1    = (const float*)d_in[4];
    const float* W2    = (const float*)d_in[5];
    const float* b2    = (const float*)d_in[6];
    const float* W3    = (const float*)d_in[7];
    const float* b3    = (const float*)d_in[8];
    const float* W_dec = (const float*)d_in[9];
    const float* b_dec = (const float*)d_in[10];
    _Float16* ws = (_Float16*)d_ws;

    prep_kernel<<<(WS_HALVES + 255) / 256, 256, 0, stream>>>(W_enc, W1, W2, W3, W_dec, ws);

    const int B = 65536;
    const int waves  = B / 16;                    // 4096 waves, 16 rows each
    const int blocks = waves / WAVES_PER_BLOCK;   // 512 blocks of 8 waves
    const size_t shmem = (size_t)TOTAL_LDS_H * sizeof(_Float16);  // 320 KB / block
    node_kernel<<<blocks, 256, shmem, stream>>>(x, ws, b_enc, b1, b2, b3, b_dec,
                                                (float*)d_out);
}
